// GaussianGraphSAGE_82377472738051
// MI455X (gfx1250) — matrix operand order repaired and checked by reading, not compiled, not run on hardware
//
#include <hip/hip_runtime.h>
#include <hip/hip_bf16.h>
#include <math.h>

#define NN 100000   // nodes
#define NE 1000000  // edges
#define NG 128      // graphs
#define DD 64       // channels
#define FCH 128
#define NC 10

typedef __attribute__((ext_vector_type(16))) _Float16 v16h;
typedef __attribute__((ext_vector_type(8)))  float    v8f;

// ---------------- utility kernels ----------------
__global__ void zero_f32(float* p, long long n) {
  long long i = (long long)blockIdx.x * blockDim.x + threadIdx.x;
  if (i < n) p[i] = 0.0f;
}

__global__ void deg_kernel(const int* __restrict__ tgt, float* __restrict__ deg) {
  int e = blockIdx.x * blockDim.x + threadIdx.x;
  if (e < NE) atomicAdd(&deg[tgt[e]], 1.0f);
}

__global__ void deginv_kernel(float* deg) {
  int n = blockIdx.x * blockDim.x + threadIdx.x;
  if (n < NN) deg[n] = 1.0f / fmaxf(deg[n], 1.0f);
}

// one thread per (edge, channel): coalesced gather of h[src], atomic scatter to agg[tgt]
__global__ void aggregate_kernel(const float* __restrict__ h, const int* __restrict__ src,
                                 const int* __restrict__ tgt, float* __restrict__ agg) {
  long long idx = (long long)blockIdx.x * blockDim.x + threadIdx.x;
  if (idx >= (long long)NE * DD) return;
  int e = (int)(idx >> 6);
  int d = (int)(idx & 63);
  int s = src[e];
  int t = tgt[e];
  atomicAdd(&agg[(long long)t * DD + d], h[(long long)s * DD + d]);
}

// ---------------- WMMA SAGE layer: out = relu(agg*deg_inv @ Wl^T + bl + h @ Wr^T) ----------------
// 16-bit A 16x32 fragment: lanes 0-15 hold K in {0..7, 16..23}, lanes 16-31 K in {8..15, 24..31}
__device__ inline v16h load_a16(const float* __restrict__ rowp, int t, int hi, float scale) {
  v16h a;
  const float* p0 = rowp + 32 * t + 8 * hi;       // K = 32t + 8hi .. +7
  const float* p1 = p0 + 16;                      // K = 32t + 16 + 8hi .. +7
#pragma unroll
  for (int j = 0; j < 8; ++j) {
    a[j]     = (_Float16)(p0[j] * scale);
    a[8 + j] = (_Float16)(p1[j] * scale);
  }
  return a;
}

// 16-bit B 32x16 fragment: lanes 0-15 hold K in {0..7, 16..23}, lanes 16-31 K in {8..15, 24..31}, N = lane%16
// B[k][n] = W[n][k]  (out = in @ W^T) -> two runs of 8 consecutive floats of one W row.
__device__ inline v16h load_b16(const float* wrow, int t, int hi) {
  v16h b;
  const float* p = wrow + 32 * t + 8 * hi;
#pragma unroll
  for (int i = 0; i < 8; ++i) { b[i] = (_Float16)p[i]; b[8 + i] = (_Float16)p[16 + i]; }
  return b;
}

#define GEMM_WAVES 8
#define GEMM_ROWS (16 * GEMM_WAVES)

__global__ void __launch_bounds__(256)
sage_gemm_kernel(const float* __restrict__ agg, const float* hin,
                 const float* __restrict__ deg_inv,
                 const float* __restrict__ Wl, const float* __restrict__ bl,
                 const float* __restrict__ Wr,
                 float* hout) {  // hout may alias hin (in-place): all A-loads precede stores
  __shared__ float sWl[DD * DD];
  __shared__ float sWr[DD * DD];
  for (int i = threadIdx.x; i < DD * DD; i += 256) {
    sWl[i] = Wl[i];
    sWr[i] = Wr[i];
  }
  __syncthreads();

  int lane = threadIdx.x & 31;
  int wave = threadIdx.x >> 5;
  int row  = lane & 15;
  int hi   = lane >> 4;
  int m0   = blockIdx.x * GEMM_ROWS + wave * 16;
  int mA   = m0 + row;
  if (mA > NN - 1) mA = NN - 1;  // clamp loads; stores guarded below (EXEC stays all-1 for WMMA)

  const float* aggRow = agg + (long long)mA * DD;
  const float* hRow   = hin + (long long)mA * DD;
  __builtin_prefetch(aggRow, 0, 0);
  __builtin_prefetch(hRow, 0, 0);
  float dinv = deg_inv[mA];

  // load all A fragments up-front (enables in-place h update)
  v16h a1_0 = load_a16(aggRow, 0, hi, dinv);
  v16h a1_1 = load_a16(aggRow, 1, hi, dinv);
  v16h a2_0 = load_a16(hRow,   0, hi, 1.0f);
  v16h a2_1 = load_a16(hRow,   1, hi, 1.0f);

#pragma unroll
  for (int n0 = 0; n0 < DD; n0 += 16) {
    const float* wlRow = &sWl[(n0 + row) * DD];
    const float* wrRow = &sWr[(n0 + row) * DD];
    v8f c = {};
    c = __builtin_amdgcn_wmma_f32_16x16x32_f16(false, a1_0, false, load_b16(wlRow, 0, hi), (short)0, c, false, false);
    c = __builtin_amdgcn_wmma_f32_16x16x32_f16(false, a1_1, false, load_b16(wlRow, 1, hi), (short)0, c, false, false);
    c = __builtin_amdgcn_wmma_f32_16x16x32_f16(false, a2_0, false, load_b16(wrRow, 0, hi), (short)0, c, false, false);
    c = __builtin_amdgcn_wmma_f32_16x16x32_f16(false, a2_1, false, load_b16(wrRow, 1, hi), (short)0, c, false, false);
    float bias = bl[n0 + row];
    // C layout: VGPR v -> M = m0 + v + 8*hi, N = n0 + row
#pragma unroll
    for (int v = 0; v < 8; ++v) {
      int m = m0 + v + 8 * hi;
      if (m < NN) {
        float o = c[v] + bias;
        hout[(long long)m * DD + n0 + row] = fmaxf(o, 0.0f);
      }
    }
  }
}

// ---------------- reparameterize + copy outputs + graph pool accumulation ----------------
__global__ void reparam_pool_kernel(const float* __restrict__ hmean, const float* __restrict__ hvar,
                                    const float* __restrict__ eps, const int* __restrict__ batch,
                                    float* __restrict__ out_mean, float* __restrict__ out_var,
                                    float* __restrict__ zg, float* __restrict__ cnt) {
  long long idx = (long long)blockIdx.x * blockDim.x + threadIdx.x;
  if (idx >= (long long)NN * DD) return;
  int n = (int)(idx >> 6);
  int d = (int)(idx & 63);
  float m  = hmean[idx];
  float lv = hvar[idx];
  out_mean[idx] = m;
  out_var[idx]  = lv;
  float z = m + eps[idx] * expf(0.5f * lv);
  int b = batch[n];
  atomicAdd(&zg[b * DD + d], z);
  if (d == 0) atomicAdd(&cnt[b], 1.0f);
}

// ---------------- MLP head ----------------
__global__ void head1_kernel(const float* __restrict__ zg, const float* __restrict__ cnt,
                             const float* __restrict__ fc1W, const float* __restrict__ fc1b,
                             float* __restrict__ h1) {
  int idx = blockIdx.x * blockDim.x + threadIdx.x;
  if (idx >= NG * FCH) return;
  int g = idx >> 7;
  int j = idx & 127;
  float ci = 1.0f / fmaxf(cnt[g], 1.0f);
  float s = fc1b[j];
#pragma unroll 8
  for (int d = 0; d < DD; ++d) s += zg[g * DD + d] * ci * fc1W[j * DD + d];
  h1[idx] = fmaxf(s, 0.0f);
}

__global__ void head2_kernel(const float* __restrict__ h1, const float* __restrict__ fc2W,
                             const float* __restrict__ fc2b, float* __restrict__ out) {
  int g = blockIdx.x * blockDim.x + threadIdx.x;
  if (g >= NG) return;
  float lg[NC];
#pragma unroll
  for (int c = 0; c < NC; ++c) {
    float s = fc2b[c];
    for (int j = 0; j < FCH; ++j) s += h1[g * FCH + j] * fc2W[c * FCH + j];
    lg[c] = s;
  }
  float mx = lg[0];
#pragma unroll
  for (int c = 1; c < NC; ++c) mx = fmaxf(mx, lg[c]);
  float se = 0.0f;
#pragma unroll
  for (int c = 0; c < NC; ++c) se += expf(lg[c] - mx);
  float lse = logf(se);
#pragma unroll
  for (int c = 0; c < NC; ++c) out[g * NC + c] = lg[c] - mx - lse;
}

// ---------------- launcher ----------------
extern "C" void kernel_launch(void* const* d_in, const int* in_sizes, int n_in,
                              void* d_out, int out_size, void* d_ws, size_t ws_size,
                              hipStream_t stream) {
  const float* x     = (const float*)d_in[0];
  const int*   edge  = (const int*)d_in[1];
  const int*   batch = (const int*)d_in[2];
  const float* eps   = (const float*)d_in[3];
  const float* mWl   = (const float*)d_in[4];
  const float* mbl   = (const float*)d_in[5];
  const float* mWr   = (const float*)d_in[6];
  const float* vWl   = (const float*)d_in[7];
  const float* vbl   = (const float*)d_in[8];
  const float* vWr   = (const float*)d_in[9];
  const float* fc1W  = (const float*)d_in[10];
  const float* fc1b  = (const float*)d_in[11];
  const float* fc2W  = (const float*)d_in[12];
  const float* fc2b  = (const float*)d_in[13];

  const int* src = edge;       // edge_index[0]
  const int* tgt = edge + NE;  // edge_index[1]

  const long long nd = (long long)NN * DD;
  float* ws  = (float*)d_ws;
  float* deg = ws;                  // NN floats (becomes deg_inv in place)
  float* agg = ws + 102400;         // NN*DD
  float* hm  = agg + nd;            // NN*DD
  float* hv  = hm + nd;             // NN*DD
  float* zg  = hv + nd;             // NG*DD
  float* cnt = zg + NG * DD;        // NG
  float* h1  = cnt + NG;            // NG*FCH

  float* out        = (float*)d_out;
  float* out_logits = out;
  float* out_mean   = out + NG * NC;
  float* out_var    = out_mean + nd;

  const int tb = 256;
  const int nBlkNodes = (NN + tb - 1) / tb;
  const int nBlkEdges = (NE + tb - 1) / tb;
  const int nBlkND    = (int)((nd + tb - 1) / tb);
  const int nBlkAgg   = (int)(((long long)NE * DD + tb - 1) / tb);
  const int nBlkGemm  = (NN + GEMM_ROWS - 1) / GEMM_ROWS;

  // degrees -> deg_inv
  zero_f32<<<nBlkNodes, tb, 0, stream>>>(deg, NN);
  deg_kernel<<<nBlkEdges, tb, 0, stream>>>(tgt, deg);
  deginv_kernel<<<nBlkNodes, tb, 0, stream>>>(deg);

  // layer 0: both branches consume x (shared aggregation)
  zero_f32<<<nBlkND, tb, 0, stream>>>(agg, nd);
  aggregate_kernel<<<nBlkAgg, tb, 0, stream>>>(x, src, tgt, agg);
  sage_gemm_kernel<<<nBlkGemm, tb, 0, stream>>>(agg, x, deg, mWl, mbl, mWr, hm);
  sage_gemm_kernel<<<nBlkGemm, tb, 0, stream>>>(agg, x, deg, vWl, vbl, vWr, hv);

  // layers 1..2 (in-place h update is safe: per-block reads precede writes)
  for (int l = 1; l < 3; ++l) {
    const int wOff = l * DD * DD, bOff = l * DD;
    zero_f32<<<nBlkND, tb, 0, stream>>>(agg, nd);
    aggregate_kernel<<<nBlkAgg, tb, 0, stream>>>(hm, src, tgt, agg);
    sage_gemm_kernel<<<nBlkGemm, tb, 0, stream>>>(agg, hm, deg, mWl + wOff, mbl + bOff, mWr + wOff, hm);
    zero_f32<<<nBlkND, tb, 0, stream>>>(agg, nd);
    aggregate_kernel<<<nBlkAgg, tb, 0, stream>>>(hv, src, tgt, agg);
    sage_gemm_kernel<<<nBlkGemm, tb, 0, stream>>>(agg, hv, deg, vWl + wOff, vbl + bOff, vWr + wOff, hv);
  }

  // reparameterize + pool + head
  zero_f32<<<(NG * DD + NG + tb - 1) / tb, tb, 0, stream>>>(zg, NG * DD + NG);
  reparam_pool_kernel<<<nBlkND, tb, 0, stream>>>(hm, hv, eps, batch, out_mean, out_var, zg, cnt);
  head1_kernel<<<(NG * FCH + tb - 1) / tb, tb, 0, stream>>>(zg, cnt, fc1W, fc1b, h1);
  head2_kernel<<<1, NG, 0, stream>>>(h1, fc2W, fc2b, out_logits);
}
